// DNCAddress_15006615733320
// MI455X (gfx1250) — matrix operand order repaired and checked by reading, not compiled, not run on hardware
//
#include <hip/hip_runtime.h>

// DNC addressing for MI455X (gfx1250), wave32.
// B=32, N=2048, W=64, C=1024. Outputs concat: [write, content, directional, allocation], each [B,N] f32.

#define Bsz 32
#define Nsz 2048
#define Wsz 64
#define Csz 1024
#define EPSF 1e-8f
#define CUT 24           // exp(-d^2/den) underflows f32 for d>15 (den<=2); 24 is safe margin
#define CHUNK_ROWS 128   // rows per sim block (TDM tile: 128x64 f32 = 32KB LDS)

typedef __bf16 bf16_t;
typedef __attribute__((ext_vector_type(16))) __bf16 v16bf;
typedef __attribute__((ext_vector_type(8)))  float  v8f;
typedef __attribute__((ext_vector_type(4)))  unsigned int v4u;
typedef __attribute__((ext_vector_type(8)))  int v8i;
typedef __attribute__((ext_vector_type(4)))  int v4i;

// Convert 8 contiguous f32 at p (16B aligned) into bf16 frag elements [o..o+7].
__device__ __forceinline__ void cvt8(v16bf& v, int o, const float* __restrict__ p) {
    const float4 x0 = ((const float4*)p)[0];
    const float4 x1 = ((const float4*)p)[1];
    v[o + 0] = (bf16_t)x0.x; v[o + 1] = (bf16_t)x0.y;
    v[o + 2] = (bf16_t)x0.z; v[o + 3] = (bf16_t)x0.w;
    v[o + 4] = (bf16_t)x1.x; v[o + 5] = (bf16_t)x1.y;
    v[o + 6] = (bf16_t)x1.z; v[o + 7] = (bf16_t)x1.w;
}

// ---------------------------------------------------------------------------
// Kernel A: k = tanh(co @ Wk^T + bk)  -> bf16 for the sim WMMA.
// D[32,64], K=1024. One 256-thread block = 8 waves; wave -> (mt,nt) 16x16 tile.
// ---------------------------------------------------------------------------
__global__ __launch_bounds__(256) void kproj_kernel(
    const float* __restrict__ co, const float* __restrict__ Wk,
    const float* __restrict__ bk, bf16_t* __restrict__ kbf)
{
    const int tid  = threadIdx.x;
    const int wave = tid >> 5;
    const int lane = tid & 31;
    const int mt = wave >> 2;      // 0..1 : batch-row tile
    const int nt = wave & 3;       // 0..3 : output-col tile
    const int lh = lane & 15;
    const int g  = lane >> 4;      // K-half selector

    const float* aRow = co + (size_t)(mt * 16 + lh) * Csz;   // A: controller row (M x K)
    const float* bRow = Wk + (size_t)(nt * 16 + lh) * Csz;   // B: Wk row n holds K-contig data

    v8f acc = {};
    for (int kk = 0; kk < Csz; kk += 32) {
        v16bf a, b;
        // 16-bit A 16x32 layout: lane group g -> K = kk + 16*(j>>3) + g*8 + (j&7)
        cvt8(a, 0, aRow + kk + g * 8);
        cvt8(a, 8, aRow + kk + g * 8 + 16);
        // 16-bit B 32x16 layout: lane group g -> K = kk + 16*(j>>3) + g*8 + (j&7), N = lane&15
        cvt8(b, 0, bRow + kk + g * 8);
        cvt8(b, 8, bRow + kk + g * 8 + 16);
        acc = __builtin_amdgcn_wmma_f32_16x16x32_bf16(
                  false, a, false, b, (short)0, acc, false, false);
    }
    // C/D layout: element r -> M = r + 8*g, N = lane&15
    const int wcol = nt * 16 + lh;
    const float bias = bk[wcol];
    for (int r = 0; r < 8; ++r) {
        const int brow = mt * 16 + g * 8 + r;
        kbf[brow * Wsz + wcol] = (bf16_t)tanhf(acc[r] + bias);
    }
}

// ---------------------------------------------------------------------------
// Kernel B: per-batch scalars: beta (softplus), sc -> den = 2*sc^2+eps,
// gate (sigmoid), and circulant row-sum normalizer S_b + eps.
// ---------------------------------------------------------------------------
__global__ __launch_bounds__(128) void scalars_kernel(
    const float* __restrict__ co,
    const float* __restrict__ Wb, const float* __restrict__ bb,
    const float* __restrict__ Ws, const float* __restrict__ bs,
    const float* __restrict__ Wg, const float* __restrict__ bg,
    float* __restrict__ betaA, float* __restrict__ denA,
    float* __restrict__ gateA, float* __restrict__ snormA)
{
    __shared__ float red[128];
    __shared__ float dots[5];
    __shared__ float denS;
    const int b = blockIdx.x, t = threadIdx.x;
    const float* row = co + (size_t)b * Csz;
    const float* wrows[5] = { Wb, Ws, Ws + Csz, Ws + 2 * Csz, Wg };

    for (int v = 0; v < 5; ++v) {
        const float* w = wrows[v];
        float p = 0.f;
        for (int c = t; c < Csz; c += 128) p += row[c] * w[c];
        red[t] = p; __syncthreads();
        for (int s = 64; s > 0; s >>= 1) { if (t < s) red[t] += red[t + s]; __syncthreads(); }
        if (t == 0) dots[v] = red[0];
        __syncthreads();
    }
    if (t == 0) {
        float xb = dots[0] + bb[0];
        betaA[b] = (xb > 20.f) ? xb : log1pf(expf(xb));
        float s0 = dots[1] + bs[0], s1 = dots[2] + bs[1], s2 = dots[3] + bs[2];
        float m = fmaxf(s0, fmaxf(s1, s2));
        float e0 = expf(s0 - m), e1 = expf(s1 - m), e2 = expf(s2 - m);
        float inv = 1.f / (e0 + e1 + e2);
        float sc = (e2 - e0) * inv;
        float d = 2.f * sc * sc + EPSF;
        denA[b] = d; denS = d;
        gateA[b] = 1.f / (1.f + expf(-(dots[4] + bg[0])));
    }
    __syncthreads();
    const float d = denS;
    float s = 0.f;
    for (int i = t; i < Nsz; i += 128) { float fi = (float)i; s += expf(-(fi * fi) / d); }
    red[t] = s; __syncthreads();
    for (int st = 64; st > 0; st >>= 1) { if (t < st) red[t] += red[t + st]; __syncthreads(); }
    if (t == 0) snormA[b] = red[0] + EPSF;
}

// ---------------------------------------------------------------------------
// Kernel C1: sim[b,n] = memory[b,n,:] . k[b,:]
// Grid = B*16 blocks; each block TDM-DMAs a 128x64 f32 tile into LDS
// (tensor_load_to_lds + s_wait_tensorcnt), then 8 waves do one 16-row WMMA
// tile each (A from LDS, B = k broadcast so every D column equals sim).
// Raw sim is written into the content output slice (overwritten by softmax).
// ---------------------------------------------------------------------------
__global__ __launch_bounds__(256) void sim_kernel(
    const float* __restrict__ mem, const bf16_t* __restrict__ kbf,
    float* __restrict__ simOut)
{
    __shared__ __align__(16) float tile[CHUNK_ROWS * Wsz];   // 32 KB
    const int bc = blockIdx.x;
    const int b  = bc >> 4;        // batch
    const int c  = bc & 15;        // 128-row chunk within batch
    const int tid = threadIdx.x;
    const int wave = tid >> 5, lane = tid & 31, lh = lane & 15, g = lane >> 4;

    const float* gsrc = mem + ((size_t)b * Nsz + (size_t)c * CHUNK_ROWS) * Wsz;

    if (tid < 32) {   // wave 0 issues the TDM DMA; TENSORcnt is per-wave
        const unsigned long long ga = (unsigned long long)(uintptr_t)gsrc;
        const unsigned ldsb = (unsigned)(uintptr_t)(void*)tile;  // LDS aperture: addr[31:0] = LDS offset
        // D# group 0: count=1 | lds_addr | global_addr[56:0] | type=2 (bits 127:126)
        v4u g0 = { 1u, ldsb, (unsigned)ga,
                   (unsigned)((ga >> 32) & 0x01FFFFFFu) | 0x80000000u };
        // D# group 1: data_size=4B, tensor_dim0=64, tensor_dim1=128,
        //             tile_dim0=64, tile_dim1=128, tensor_dim0_stride=64
        v8i g1 = { (int)0x00020000,          // data_size=2 (4 bytes) at bits[17:16]
                   (int)(64u << 16),         // tensor_dim0[15:0] at bits[63:48]
                   (int)(128u << 16),        // tensor_dim1[15:0] at bits[111:80]->word2[31:16]
                   (int)(64u << 16),         // tile_dim0 at bits[127:112]
                   (int)128,                 // tile_dim1 at bits[143:128]
                   (int)64,                  // tensor_dim0_stride at bits[207:160]
                   0, 0 };
        v4i gz = {};                          // groups 2/3: 2-D tile, unused dims = 0
#if defined(__clang_major__) && (__clang_major__ >= 23)
        v8i gz8 = {};
        __builtin_amdgcn_tensor_load_to_lds(g0, g1, gz, gz, gz8, 0);
#else
        __builtin_amdgcn_tensor_load_to_lds(g0, g1, gz, gz, 0);
#endif
        __builtin_amdgcn_s_wait_tensorcnt(0);
    }
    __syncthreads();

    // Broadcast B frags: lane group g holds k[ks + g*16 + j]
    const v16bf bf0 = *(const v16bf*)(kbf + (size_t)b * Wsz + g * 16);
    const v16bf bf1 = *(const v16bf*)(kbf + (size_t)b * Wsz + 32 + g * 16);

    const int n0 = wave * 16;                         // one 16-row tile per wave
    const float* row = tile + (size_t)(n0 + lh) * Wsz;
    v8f acc = {};
    v16bf a;
    cvt8(a, 0, row + g * 16);                         // K = g*16 + j
    cvt8(a, 8, row + g * 16 + 8);                     // K = g*16 + 8 + j
    acc = __builtin_amdgcn_wmma_f32_16x16x32_bf16(
              false, a, false, bf0, (short)0, acc, false, false);
    cvt8(a, 0, row + 32 + g * 16);
    cvt8(a, 8, row + 40 + g * 16);
    acc = __builtin_amdgcn_wmma_f32_16x16x32_bf16(
              false, a, false, bf1, (short)0, acc, false, false);

    if (lh == 0) {   // D columns identical; lanes 0 & 16 own the two M halves
        const size_t base = (size_t)b * Nsz + (size_t)c * CHUNK_ROWS + n0 + g * 8;
        for (int r = 0; r < 8; ++r) simOut[base + r] = acc[r];
    }
}

// ---------------------------------------------------------------------------
// Kernel C2: in-place softmax over sim (scaled by beta) -> content weights.
// One block per batch; touches only 8KB/batch.
// ---------------------------------------------------------------------------
__global__ __launch_bounds__(256) void content_softmax_kernel(
    const float* __restrict__ betaA, float* __restrict__ content)
{
    __shared__ float sim[Nsz];
    __shared__ float red[256];
    const int b = blockIdx.x, tid = threadIdx.x;
    const float beta = betaA[b];
    for (int i = tid; i < Nsz; i += 256) sim[i] = beta * content[(size_t)b * Nsz + i];
    __syncthreads();
    float m = -1e30f;
    for (int i = tid; i < Nsz; i += 256) m = fmaxf(m, sim[i]);
    red[tid] = m; __syncthreads();
    for (int s = 128; s > 0; s >>= 1) { if (tid < s) red[tid] = fmaxf(red[tid], red[tid + s]); __syncthreads(); }
    m = red[0]; __syncthreads();
    float ssum = 0.f;
    for (int i = tid; i < Nsz; i += 256) ssum += expf(sim[i] - m);
    red[tid] = ssum; __syncthreads();
    for (int s = 128; s > 0; s >>= 1) { if (tid < s) red[tid] += red[tid + s]; __syncthreads(); }
    const float inv = 1.f / red[0];
    for (int i = tid; i < Nsz; i += 256)
        content[(size_t)b * Nsz + i] = expf(sim[i] - m) * inv;
}

// ---------------------------------------------------------------------------
// Kernel D: directional weights. Circulant kernel => banded circular
// correlation with taps g(d)=exp(-d^2/den), support d<=CUT (f32 underflow),
// normalized by the constant row sum S_b + eps.
// ---------------------------------------------------------------------------
__global__ __launch_bounds__(256) void directional_kernel(
    const float* __restrict__ prevRead, const float* __restrict__ denA,
    const float* __restrict__ snormA, float* __restrict__ dirOut)
{
    __shared__ float pr[Nsz];
    __shared__ float gk[CUT + 1];
    const int b = blockIdx.x, t = threadIdx.x;
    const float den  = denA[b];
    const float invS = 1.f / snormA[b];
    for (int i = t; i < Nsz; i += 256) pr[i] = prevRead[(size_t)b * Nsz + i];
    if (t <= CUT) { float fd = (float)t; gk[t] = expf(-(fd * fd) / den); }
    __syncthreads();
    for (int m = t; m < Nsz; m += 256) {
        const int base = (m - Nsz / 2) & (Nsz - 1);   // n index at d=0
        float s = 0.f;
        #pragma unroll
        for (int d = 0; d <= CUT; ++d) s += gk[d] * pr[(base + d) & (Nsz - 1)];
        dirOut[(size_t)b * Nsz + m] = s * invS;
    }
}

// ---------------------------------------------------------------------------
// Kernel E: allocation weights. LDS bitonic sort of (usage, idx) ascending
// (stable via idx tie-break), multiplicative chunked scan, scatter-back.
// ---------------------------------------------------------------------------
__global__ __launch_bounds__(256) void alloc_kernel(
    const float* __restrict__ usage, float* __restrict__ allocOut)
{
    __shared__ float vals[Nsz];
    __shared__ int   idxs[Nsz];
    __shared__ float scan[256];
    const int b = blockIdx.x, t = threadIdx.x;
    for (int i = t; i < Nsz; i += 256) { vals[i] = usage[(size_t)b * Nsz + i]; idxs[i] = i; }
    __syncthreads();

    for (int k = 2; k <= Nsz; k <<= 1) {
        for (int j = k >> 1; j > 0; j >>= 1) {
            for (int i = t; i < Nsz; i += 256) {
                const int p = i ^ j;
                if (p > i) {
                    const bool up = ((i & k) == 0);
                    float vi = vals[i], vp = vals[p];
                    int   ii = idxs[i], ip = idxs[p];
                    const bool gt = (vi > vp) || (vi == vp && ii > ip);
                    if (gt == up) { vals[i] = vp; vals[p] = vi; idxs[i] = ip; idxs[p] = ii; }
                }
            }
            __syncthreads();
        }
    }

    // inclusive cumprod of (1 - sorted_usage): 8-wide chunks + Hillis-Steele product scan
    float local[8];
    const int base = t * 8;
    float prod = 1.f;
    #pragma unroll
    for (int e = 0; e < 8; ++e) { prod *= (1.f - vals[base + e]); local[e] = prod; }
    scan[t] = prod;
    __syncthreads();
    for (int off = 1; off < 256; off <<= 1) {
        float v = scan[t];
        if (t >= off) v *= scan[t - off];
        __syncthreads();
        scan[t] = v;
        __syncthreads();
    }
    const float prefix = (t == 0) ? 1.f : scan[t - 1];
    #pragma unroll
    for (int e = 0; e < 8; ++e)
        allocOut[(size_t)b * Nsz + idxs[base + e]] = prefix * local[e];
}

// ---------------------------------------------------------------------------
// Kernel F: write = gate * (0.5*content + 0.5*directional*allocation)
// ---------------------------------------------------------------------------
__global__ __launch_bounds__(256) void combine_kernel(
    const float* __restrict__ content, const float* __restrict__ dir,
    const float* __restrict__ alloc,   const float* __restrict__ gateA,
    float* __restrict__ ww)
{
    const int i = blockIdx.x * 256 + threadIdx.x;
    const int b = i >> 11;
    ww[i] = gateA[b] * (0.5f * content[i] + 0.5f * dir[i] * alloc[i]);
}

extern "C" void kernel_launch(void* const* d_in, const int* in_sizes, int n_in,
                              void* d_out, int out_size, void* d_ws, size_t ws_size,
                              hipStream_t stream)
{
    (void)in_sizes; (void)n_in; (void)out_size; (void)ws_size;
    const float* co    = (const float*)d_in[0];
    const float* prw   = (const float*)d_in[1];
    /* d_in[2] prev_write_weight: unused by the reference */
    const float* mem   = (const float*)d_in[3];
    const float* usage = (const float*)d_in[4];
    const float* Wk    = (const float*)d_in[5];
    const float* bk    = (const float*)d_in[6];
    const float* Wb    = (const float*)d_in[7];
    const float* bb    = (const float*)d_in[8];
    const float* Ws    = (const float*)d_in[9];
    const float* bs    = (const float*)d_in[10];
    const float* Wg    = (const float*)d_in[11];
    const float* bg    = (const float*)d_in[12];

    float* out = (float*)d_out;
    const int BN = Bsz * Nsz;
    float* ww = out;
    float* cw = out + BN;        // holds raw sim, then content (in-place softmax)
    float* dw = out + 2 * BN;
    float* aw = out + 3 * BN;

    char* ws = (char*)d_ws;
    bf16_t* kbf  = (bf16_t*)ws;             // 32*64 bf16 = 4096 B
    float* betaA = (float*)(ws + 4096);     // 32 f32 each
    float* denA  = betaA + 32;
    float* gateA = betaA + 64;
    float* snorm = betaA + 96;

    kproj_kernel          <<<1,         256, 0, stream>>>(co, Wk, bk, kbf);
    scalars_kernel        <<<Bsz,       128, 0, stream>>>(co, Wb, bb, Ws, bs, Wg, bg,
                                                          betaA, denA, gateA, snorm);
    alloc_kernel          <<<Bsz,       256, 0, stream>>>(usage, aw);
    sim_kernel            <<<Bsz * 16,  256, 0, stream>>>(mem, kbf, cw);
    content_softmax_kernel<<<Bsz,       256, 0, stream>>>(betaA, cw);
    directional_kernel    <<<Bsz,       256, 0, stream>>>(prw, denA, snorm, dw);
    combine_kernel        <<<BN / 256,  256, 0, stream>>>(cw, dw, aw, gateA, ww);
}
